// Constructor_90271622627605
// MI455X (gfx1250) — matrix operand order repaired and checked by reading, not compiled, not run on hardware
//
#include <hip/hip_runtime.h>
#include <stdint.h>

// ---- model dims (fixed by the reference) ----
#define Cdim 1024
#define Hn   16
#define HSd  64
#define Vv   32000
#define Tt   1024
#define Bb   2
#define NLn  6

typedef __attribute__((ext_vector_type(16))) __bf16 v16bf;
typedef __attribute__((ext_vector_type(2)))  __bf16 v2bf;
typedef __attribute__((ext_vector_type(8)))  float  v8f;

union Frag16 {
  v16bf v;
  unsigned short us[16];
  unsigned int w[8];
  uint4 q[2];
};

__device__ __forceinline__ unsigned short f2bf(float f) {
  unsigned int u = __float_as_uint(f);
  u += 0x7FFFu + ((u >> 16) & 1u);   // round-to-nearest-even
  return (unsigned short)(u >> 16);
}

// two fp32 -> packed bf16 dword (RTNE). Prefer hw packed cvt; else round both
// and merge the two high halves with a single v_perm_b32.
__device__ __forceinline__ unsigned int pk2bf(float a, float b) {
#if __has_builtin(__builtin_amdgcn_cvt_pk_bf16_f32)
  union { v2bf v; unsigned int u; } cv;
  cv.v = __builtin_amdgcn_cvt_pk_bf16_f32(a, b);
  return cv.u;
#else
  unsigned int ua = __float_as_uint(a), ub = __float_as_uint(b);
  ua += 0x7FFFu + ((ua >> 16) & 1u);
  ub += 0x7FFFu + ((ub >> 16) & 1u);
#if __has_builtin(__builtin_amdgcn_perm)
  return __builtin_amdgcn_perm(ub, ua, 0x07060302u);  // {ub[31:16], ua[31:16]}
#else
  return (ua >> 16) | (ub & 0xFFFF0000u);
#endif
#endif
}

// =====================================================================
// GEMM: C[M,N] = A16[M,K](bf16) * B[K,N](fp32 weights) (+bias)(+relu).
// Outputs fp32 (Cf) and/or bf16 (C16) - whichever is non-null.
// blayout: 0 = row-major [K,N]; 1 = head-blocked [H,K,64] (n -> h=n>>6,d=n&63).
// Block tile 128x128, K-step 32, 8 waves each computing 64x32.
// =====================================================================
__global__ __launch_bounds__(256)
void gemm_wmma_kernel(const unsigned short* __restrict__ A16,
                      const float* __restrict__ Bm,
                      const float* __restrict__ bias,
                      float* __restrict__ Cf, unsigned short* __restrict__ C16,
                      int M, int N, int K, int blayout, int relu)
{
  __shared__ unsigned short As[128][40];   // [m][k], pad 8 halves
  __shared__ unsigned short Bs[128][40];   // [n][k] (transposed), pad 8 halves

  const int tid  = threadIdx.x;
  const int lane = tid & 31, wave = tid >> 5;
  const int l = lane & 15, hi = lane >> 4;
  const int m0 = blockIdx.y * 128, n0 = blockIdx.x * 128;
  const int wm = (wave >> 2) * 64, wn = (wave & 3) * 32;

  v8f acc[4][2] = {};

  const int arow = tid >> 1, akk = (tid & 1) * 16;   // A loader: 2 thr/row
  const int bkp = (tid & 15) * 2;                    // B loader: k pair 0..30
  const int bnb = (tid >> 4) * 8;                    // B loader: 8 cols/thread

  for (int k0 = 0; k0 < K; k0 += 32) {
    // ---- stage A tile: bf16 in memory -> raw 16B copies, zero VALU ----
    {
      const unsigned short* ap = A16 + (size_t)(m0 + arow) * K + k0 + akk;
      uint4 v0 = *(const uint4*)(ap);
      uint4 v1 = *(const uint4*)(ap + 8);
      *(uint4*)&As[arow][akk]     = v0;
      *(uint4*)&As[arow][akk + 8] = v1;
    }
    // ---- stage B tile (fp32 weights) transposed, k-pairs packed ----
    if (blayout == 0) {
      const float* bp0 = Bm + (size_t)(k0 + bkp) * N + n0 + bnb;
      const float* bp1 = bp0 + N;
      if (k0 + 32 < K) __builtin_prefetch(bp0 + (size_t)32 * N, 0, 1);
      float4 r0 = *(const float4*)(bp0);
      float4 r1 = *(const float4*)(bp0 + 4);
      float4 s0 = *(const float4*)(bp1);
      float4 s1 = *(const float4*)(bp1 + 4);
      *(unsigned int*)&Bs[bnb + 0][bkp] = pk2bf(r0.x, s0.x);
      *(unsigned int*)&Bs[bnb + 1][bkp] = pk2bf(r0.y, s0.y);
      *(unsigned int*)&Bs[bnb + 2][bkp] = pk2bf(r0.z, s0.z);
      *(unsigned int*)&Bs[bnb + 3][bkp] = pk2bf(r0.w, s0.w);
      *(unsigned int*)&Bs[bnb + 4][bkp] = pk2bf(r1.x, s1.x);
      *(unsigned int*)&Bs[bnb + 5][bkp] = pk2bf(r1.y, s1.y);
      *(unsigned int*)&Bs[bnb + 6][bkp] = pk2bf(r1.z, s1.z);
      *(unsigned int*)&Bs[bnb + 7][bkp] = pk2bf(r1.w, s1.w);
    } else {
      #pragma unroll
      for (int i = 0; i < 8; ++i) {
        const int gn = n0 + bnb + i;
        const size_t base = ((size_t)(gn >> 6) * K + (k0 + bkp)) * 64 + (gn & 63);
        *(unsigned int*)&Bs[bnb + i][bkp] = pk2bf(Bm[base], Bm[base + 64]);
      }
    }
    __syncthreads();

    // ---- fragments: exact CDNA5 bf16 WMMA lane layouts ----
    Frag16 af[4], bfr[2];
    #pragma unroll
    for (int mi = 0; mi < 4; ++mi) {
      // A 16x32: lanes 0-15 row=l K={0..7,16..23}; lanes 16-31 K+=8
      const unsigned short* p = &As[wm + mi * 16 + l][0];
      af[mi].q[0] = *(const uint4*)(p + hi * 8);
      af[mi].q[1] = *(const uint4*)(p + 16 + hi * 8);
    }
    #pragma unroll
    for (int ni = 0; ni < 2; ++ni) {
      // B 32x16: lane col n=l; lanes 0-15 K={0..7,16..23}; lanes 16-31 K+=8
      const unsigned short* p = &Bs[wn + ni * 16 + l][0];
      bfr[ni].q[0] = *(const uint4*)(p + hi * 8);
      bfr[ni].q[1] = *(const uint4*)(p + 16 + hi * 8);
    }
    #pragma unroll
    for (int mi = 0; mi < 4; ++mi)
      #pragma unroll
      for (int ni = 0; ni < 2; ++ni)
        acc[mi][ni] = __builtin_amdgcn_wmma_f32_16x16x32_bf16(
            false, af[mi].v, false, bfr[ni].v, (short)0, acc[mi][ni], false, false);
    __syncthreads();
  }

  // ---- epilogue: C layout (vgpr r, lane) -> (M = hi*8+r, N = l) ----
  #pragma unroll
  for (int mi = 0; mi < 4; ++mi)
    #pragma unroll
    for (int ni = 0; ni < 2; ++ni) {
      const int gcol = n0 + wn + ni * 16 + l;
      const float bv = bias ? bias[gcol] : 0.f;
      #pragma unroll
      for (int r = 0; r < 8; ++r) {
        const int grow = m0 + wm + mi * 16 + hi * 8 + r;
        float vv = acc[mi][ni][r] + bv;
        if (relu) vv = fmaxf(vv, 0.f);
        if (Cf)  Cf[(size_t)grow * N + gcol] = vv;
        if (C16) C16[(size_t)grow * N + gcol] = f2bf(vv);
      }
    }
}

// =====================================================================
// Flash attention over bf16 Q/K/V [B, T, H*64] -> bf16 O.
// grid (Tq/64, H, B), 4 waves/block, 16 q-rows/wave, 32-key tiles,
// online softmax; [T,T] scores never touch HBM.
// =====================================================================
__global__ __launch_bounds__(128)
void flash_attn_kernel(const unsigned short* __restrict__ Q,
                       const unsigned short* __restrict__ Kc,
                       const unsigned short* __restrict__ Vc,
                       unsigned short* __restrict__ O,
                       const int* __restrict__ mask, int Tq, int Tk, int causal)
{
  __shared__ unsigned short Pst[4][16][40];  // per-wave P staging (C->A relayout)

  const int lane = threadIdx.x & 31, wave = threadIdx.x >> 5;
  const int l = lane & 15, hi = lane >> 4;
  const int h = blockIdx.y, b = blockIdx.z;
  const int qbase = blockIdx.x * 64 + wave * 16;
  const float scale = 0.125f;  // 1/sqrt(64)

  // Q as two 16x32 A-fragments (d-chunks): raw 16B loads, no conversion
  Frag16 qa[2];
  {
    const unsigned short* qp = Q + ((size_t)(b * Tq + qbase + l)) * Cdim + h * HSd;
    #pragma unroll
    for (int dc = 0; dc < 2; ++dc) {
      qa[dc].q[0] = *(const uint4*)(qp + dc * 32 + hi * 8);
      qa[dc].q[1] = *(const uint4*)(qp + dc * 32 + 16 + hi * 8);
    }
  }

  float mrow[8], lrow[8];
  #pragma unroll
  for (int r = 0; r < 8; ++r) { mrow[r] = -1e30f; lrow[r] = 0.f; }
  v8f oacc[4] = {};

  for (int kt = 0; kt < Tk; kt += 32) {
    if (causal && kt > qbase + 15) break;  // wave-uniform: EXEC stays full

    // K^T as B-fragments: frag[hh][dc], lane col = key, K-index = d (table order)
    Frag16 kb[2][2];
    #pragma unroll
    for (int hh = 0; hh < 2; ++hh) {
      const unsigned short* kp =
          Kc + ((size_t)(b * Tk + kt + hh * 16 + l)) * Cdim + h * HSd + hi * 8;
      #pragma unroll
      for (int dc = 0; dc < 2; ++dc) {
        kb[hh][dc].q[0] = *(const uint4*)(kp + dc * 32);
        kb[hh][dc].q[1] = *(const uint4*)(kp + dc * 32 + 16);
      }
    }

    // S = Q K^T  (2 halves x 2 d-chunks = 4 WMMAs)
    v8f s[2];
    #pragma unroll
    for (int hh = 0; hh < 2; ++hh) {
      v8f c = {};
      c = __builtin_amdgcn_wmma_f32_16x16x32_bf16(false, qa[0].v, false, kb[hh][0].v, (short)0, c, false, false);
      c = __builtin_amdgcn_wmma_f32_16x16x32_bf16(false, qa[1].v, false, kb[hh][1].v, (short)0, c, false, false);
      s[hh] = c;
    }

    // scale + causal/key masking (C layout: row = qbase+hi*8+r, key = kt+hh*16+l)
    #pragma unroll
    for (int hh = 0; hh < 2; ++hh) {
      const int kg = kt + hh * 16 + l;
      const bool mz = mask && (mask[b * Tk + kg] == 0);
      #pragma unroll
      for (int r = 0; r < 8; ++r) {
        const int rg = qbase + hi * 8 + r;
        const float sv = s[hh][r] * scale;
        const bool bad = mz || (causal && kg > rg);
        s[hh][r] = bad ? -1e30f : sv;
      }
    }

    // online softmax: per-row reduce across the 16 lanes of the row's group
    #pragma unroll
    for (int r = 0; r < 8; ++r) {
      float t = fmaxf(s[0][r], s[1][r]);
      #pragma unroll
      for (int off = 1; off < 16; off <<= 1) t = fmaxf(t, __shfl_xor(t, off, 32));
      const float mn = fmaxf(mrow[r], t);
      const float p0 = __expf(s[0][r] - mn);
      const float p1 = __expf(s[1][r] - mn);
      float ps = p0 + p1;
      #pragma unroll
      for (int off = 1; off < 16; off <<= 1) ps += __shfl_xor(ps, off, 32);
      const float al = __expf(mrow[r] - mn);
      lrow[r] = lrow[r] * al + ps;
      mrow[r] = mn;
      #pragma unroll
      for (int dc = 0; dc < 4; ++dc) oacc[dc][r] *= al;
      Pst[wave][hi * 8 + r][l]      = f2bf(p0);
      Pst[wave][hi * 8 + r][16 + l] = f2bf(p1);
    }
    asm volatile("s_wait_dscnt 0" ::: "memory");  // wave-local LDS RAW fence

    // P: C-layout -> A-fragment (16 rows x 32 keys) via LDS
    Frag16 pf;
    {
      const unsigned short* p = &Pst[wave][l][0];
      pf.q[0] = *(const uint4*)(p + hi * 8);
      pf.q[1] = *(const uint4*)(p + 16 + hi * 8);
    }

    // O += P V  (4 d-chunks of 16 output cols); V reads are raw u16
    #pragma unroll
    for (int dc = 0; dc < 4; ++dc) {
      Frag16 vf;
      #pragma unroll
      for (int j = 0; j < 16; ++j) {
        const size_t kbase =
            ((size_t)(b * Tk + kt + hi * 8 + j + (j & 8))) * Cdim + h * HSd + dc * 16 + l;
        vf.us[j] = Vc[kbase];
      }
      oacc[dc] = __builtin_amdgcn_wmma_f32_16x16x32_bf16(
          false, pf.v, false, vf.v, (short)0, oacc[dc], false, false);
    }
  }

  #pragma unroll
  for (int dc = 0; dc < 4; ++dc)
    #pragma unroll
    for (int r = 0; r < 8; ++r) {
      const int rg = qbase + hi * 8 + r;
      O[((size_t)(b * Tq + rg)) * Cdim + h * HSd + dc * 16 + l] =
          f2bf(oacc[dc][r] / lrow[r]);
    }
}

// =====================================================================
// Fused residual add + LayerNorm; writes fp32 (residual chain) and
// packed bf16 (next GEMM's A operand). 1 row/block.
// =====================================================================
__global__ __launch_bounds__(256)
void add_ln_kernel(const float* __restrict__ a, const float* __restrict__ res,
                   const float* __restrict__ g, const float* __restrict__ bb,
                   float* __restrict__ out, unsigned short* __restrict__ out16)
{
  __shared__ float rs[16];
  const int row = blockIdx.x, tid = threadIdx.x;
  const int wave = tid >> 5, lane = tid & 31;
  float x[4]; float s = 0.f, s2 = 0.f;
  #pragma unroll
  for (int i = 0; i < 4; ++i) {
    const int c = tid * 4 + i;
    float v = a[(size_t)row * Cdim + c];
    if (res) v += res[(size_t)row * Cdim + c];
    x[i] = v; s += v; s2 += v * v;
  }
  #pragma unroll
  for (int off = 1; off < 32; off <<= 1) {
    s += __shfl_xor(s, off, 32); s2 += __shfl_xor(s2, off, 32);
  }
  if (lane == 0) { rs[wave] = s; rs[8 + wave] = s2; }
  __syncthreads();
  if (wave == 0) {
    float a0 = (lane < 8) ? rs[lane] : 0.f;
    float a1 = (lane < 8) ? rs[8 + lane] : 0.f;
    #pragma unroll
    for (int off = 1; off < 8; off <<= 1) {
      a0 += __shfl_xor(a0, off, 32); a1 += __shfl_xor(a1, off, 32);
    }
    if (lane == 0) { rs[0] = a0; rs[1] = a1; }
  }
  __syncthreads();
  const float mean = rs[0] * (1.f / Cdim);
  const float var  = rs[1] * (1.f / Cdim) - mean * mean;
  const float inv  = rsqrtf(var + 1e-5f);
  float y[4];
  #pragma unroll
  for (int i = 0; i < 4; ++i) {
    const int c = tid * 4 + i;
    y[i] = (x[i] - mean) * inv * g[c] + bb[c];
    if (out) out[(size_t)row * Cdim + c] = y[i];
  }
  if (out16) {
    unsigned int* p = (unsigned int*)&out16[(size_t)row * Cdim + tid * 4];
    p[0] = pk2bf(y[0], y[1]);
    p[1] = pk2bf(y[2], y[3]);
  }
}

// =====================================================================
// Token + positional embedding gather; fp32 + bf16 dual output
// =====================================================================
__global__ __launch_bounds__(256)
void embed_kernel(const float* __restrict__ tok, const float* __restrict__ pos,
                  const int* __restrict__ ids, float* __restrict__ out,
                  unsigned short* __restrict__ out16, int T)
{
  const int row = blockIdx.x;          // b*T + t
  const int t = row % T;
  const int id = ids[row];
  const int c = threadIdx.x * 4;
  float4 tk = *(const float4*)&tok[(size_t)id * Cdim + c];
  float4 ps = *(const float4*)&pos[(size_t)t * Cdim + c];
  float4 v = make_float4(tk.x + ps.x, tk.y + ps.y, tk.z + ps.z, tk.w + ps.w);
  *(float4*)&out[(size_t)row * Cdim + c] = v;
  unsigned int* p = (unsigned int*)&out16[(size_t)row * Cdim + c];
  p[0] = pk2bf(v.x, v.y);
  p[1] = pk2bf(v.z, v.w);
}

// =====================================================================
// Host-side orchestration
// =====================================================================
extern "C" void kernel_launch(void* const* d_in, const int* in_sizes, int n_in,
                              void* d_out, int out_size, void* d_ws, size_t ws_size,
                              hipStream_t stream)
{
  (void)in_sizes; (void)n_in; (void)out_size; (void)ws_size;

  struct MHAP { const float *wk, *wq, *wv, *wo, *bo; };
  struct FFNP { const float *w1, *b1, *w2, *b2; };
  struct LNP  { const float *g, *b; };
  struct EncB { MHAP sa; FFNP ff; LNP ln1, ln2; };
  struct DecB { MHAP sa, ca; FFNP ff; LNP ln1, ln2, ln3; };

  int idx = 0;
  auto F = [&]() { return (const float*)d_in[idx++]; };

  // setup_inputs() insertion order
  const float* dec_tok = F();
  const float* dec_pos = F();
  const float* enc_tok = F();
  const float* enc_pos = F();
  EncB enc[NLn]; DecB dec[NLn];
  auto rd_mha = [&](MHAP& m) { m.wk = F(); m.wq = F(); m.wv = F(); m.wo = F(); m.bo = F(); };
  auto rd_ffn = [&](FFNP& f) { f.w1 = F(); f.b1 = F(); f.w2 = F(); f.b2 = F(); };
  auto rd_ln  = [&](LNP& p)  { p.g = F(); p.b = F(); };
  for (int i = 0; i < NLn; ++i) {
    rd_mha(enc[i].sa); rd_ffn(enc[i].ff); rd_ln(enc[i].ln1); rd_ln(enc[i].ln2);
  }
  for (int i = 0; i < NLn; ++i) {
    rd_mha(dec[i].sa); rd_mha(dec[i].ca); rd_ffn(dec[i].ff);
    rd_ln(dec[i].ln1); rd_ln(dec[i].ln2); rd_ln(dec[i].ln3);
  }
  const float* ln_g  = F();
  const float* ln_b  = F();
  const float* lid_w = F();
  const float* lid_b = F();
  const int* dec_in   = (const int*)d_in[idx++];
  const int* dec_mask = (const int*)d_in[idx++];
  const int* enc_in   = (const int*)d_in[idx++];
  const int* enc_mask = (const int*)d_in[idx++];

  // workspace: fp32 {xe,xd,t1} + bf16 {xe16,xd16,q,k,v,ab,hid} = 64 MB
  const size_t BT = (size_t)Bb * Tt;
  float* xe = (float*)d_ws;
  float* xd = xe + BT * Cdim;
  float* t1 = xd + BT * Cdim;
  unsigned short* xe16 = (unsigned short*)(t1 + BT * Cdim);
  unsigned short* xd16 = xe16 + BT * Cdim;
  unsigned short* qb   = xd16 + BT * Cdim;
  unsigned short* kb   = qb + BT * Cdim;
  unsigned short* vb   = kb + BT * Cdim;
  unsigned short* ab   = vb + BT * Cdim;
  unsigned short* hid  = ab + BT * Cdim;   // BT * 4C halves

  auto gemm = [&](const unsigned short* A16, const float* Bm, const float* bias,
                  float* Cf, unsigned short* C16, int M, int N, int K,
                  int blayout, int relu) {
    dim3 grid(N / 128, M / 128);
    gemm_wmma_kernel<<<grid, 256, 0, stream>>>(A16, Bm, bias, Cf, C16,
                                               M, N, K, blayout, relu);
  };
  auto attn = [&](const unsigned short* q, const unsigned short* k,
                  const unsigned short* v, unsigned short* o,
                  const int* mk, int causal) {
    dim3 grid(Tt / 64, Hn, Bb);
    flash_attn_kernel<<<grid, 128, 0, stream>>>(q, k, v, o, mk, Tt, Tt, causal);
  };
  auto lnorm = [&](const float* a, const float* res, LNP p,
                   float* o, unsigned short* o16) {
    add_ln_kernel<<<(int)BT, 256, 0, stream>>>(a, res, p.g, p.b, o, o16);
  };

  const int M = (int)BT;

  // ---- embeddings (fp32 residual + bf16 GEMM operand) ----
  embed_kernel<<<M, 256, 0, stream>>>(enc_tok, enc_pos, enc_in, xe, xe16, Tt);
  embed_kernel<<<M, 256, 0, stream>>>(dec_tok, dec_pos, dec_in, xd, xd16, Tt);

  // ---- encoder ----
  for (int i = 0; i < NLn; ++i) {
    const EncB& L = enc[i];
    gemm(xe16, L.sa.wq, nullptr, nullptr, qb, M, Cdim, Cdim, 1, 0);
    gemm(xe16, L.sa.wk, nullptr, nullptr, kb, M, Cdim, Cdim, 1, 0);
    gemm(xe16, L.sa.wv, nullptr, nullptr, vb, M, Cdim, Cdim, 1, 0);
    attn(qb, kb, vb, ab, enc_mask, 0);
    gemm(ab, L.sa.wo, L.sa.bo, t1, nullptr, M, Cdim, Cdim, 0, 0);
    lnorm(t1, xe, L.ln1, xe, xe16);
    gemm(xe16, L.ff.w1, L.ff.b1, nullptr, hid, M, 4 * Cdim, Cdim, 0, 1);
    gemm(hid, L.ff.w2, L.ff.b2, t1, nullptr, M, Cdim, 4 * Cdim, 0, 0);
    lnorm(t1, xe, L.ln2, xe, xe16);
  }

  // ---- decoder ----
  for (int i = 0; i < NLn; ++i) {
    const DecB& L = dec[i];
    // causal self-attention
    gemm(xd16, L.sa.wq, nullptr, nullptr, qb, M, Cdim, Cdim, 1, 0);
    gemm(xd16, L.sa.wk, nullptr, nullptr, kb, M, Cdim, Cdim, 1, 0);
    gemm(xd16, L.sa.wv, nullptr, nullptr, vb, M, Cdim, Cdim, 1, 0);
    attn(qb, kb, vb, ab, dec_mask, 1);
    gemm(ab, L.sa.wo, L.sa.bo, t1, nullptr, M, Cdim, Cdim, 0, 0);
    lnorm(t1, xd, L.ln1, xd, xd16);
    // cross-attention (q from decoder, k/v from encoder output)
    gemm(xd16, L.ca.wq, nullptr, nullptr, qb, M, Cdim, Cdim, 1, 0);
    gemm(xe16, L.ca.wk, nullptr, nullptr, kb, M, Cdim, Cdim, 1, 0);
    gemm(xe16, L.ca.wv, nullptr, nullptr, vb, M, Cdim, Cdim, 1, 0);
    attn(qb, kb, vb, ab, enc_mask, 0);
    gemm(ab, L.ca.wo, L.ca.bo, t1, nullptr, M, Cdim, Cdim, 0, 0);
    lnorm(t1, xd, L.ln3, xd, xd16);
    // FFN
    gemm(xd16, L.ff.w1, L.ff.b1, nullptr, hid, M, 4 * Cdim, Cdim, 0, 1);
    gemm(hid, L.ff.w2, L.ff.b2, t1, nullptr, M, Cdim, 4 * Cdim, 0, 0);
    lnorm(t1, xd, L.ln2, xd, xd16);
  }

  // ---- final LN + LM head (fp32 logits to d_out) ----
  LNP fin{ln_g, ln_b};
  lnorm(xd, nullptr, fin, nullptr, xd16);
  gemm(xd16, lid_w, lid_b, (float*)d_out, nullptr, M, Vv, Cdim, 0, 0);
}
